// WavetableOscillator_30081950941724
// MI455X (gfx1250) — matrix operand order repaired and checked by reading, not compiled, not run on hardware
//
#include <hip/hip_runtime.h>

// ---------------------------------------------------------------------------
// CDNA5 / gfx1250 wavetable oscillator
//   GEMM1: [W_amp;W_phase](257x256) x z_b(256x2000)  -> fused exp/mask/sincos
//   GEMM2: irfft as fixed basis F(512x514) x [Re;Im](514x2000) -> wt
//   scan + gather/interp for the oscillator
// Matrix work on v_wmma_f32_16x16x32_f16 (f16 in, f32 accumulate).
// ---------------------------------------------------------------------------

typedef __attribute__((ext_vector_type(16))) _Float16 v16h;
typedef __attribute__((ext_vector_type(8)))  float    v8f;

#define B_   16
#define C_   256
#define T_   2000
#define NB_  257          // harmonics bins
#define NS_  512          // wavetable length
#define MP1  272          // NB_ padded to 17 * 16
#define KP2  544          // 2*NB_=514 padded to 17 * 32
#define HOP  256
#define SR_  24000.0f
#define L_   (T_ * HOP)   // 512000 samples per batch

// ---- WMMA fragment loaders (layouts per CDNA5 ISA 7.12.2) -------

// A: 16x32 f16 tile from row-major [M][K] matrix, leading dim lda.
// lane<16 : M=lane,    halves0-7 -> K=kBase+0..7,  halves8-15 -> K=kBase+16..23
// lane>=16: M=lane-16, halves0-7 -> K=kBase+8..15, halves8-15 -> K=kBase+24..31
__device__ __forceinline__ v16h frag_a(const _Float16* A, int lda, int mBase,
                                       int kBase, int lane) {
  int m = mBase + (lane & 15);
  const _Float16* p = A + (size_t)m * lda + kBase + ((lane & 16) ? 8 : 0);
  union { uint4 u[2]; v16h v; } r;
  r.u[0] = *reinterpret_cast<const uint4*>(p);       // 8 halves (16B)
  r.u[1] = *reinterpret_cast<const uint4*>(p + 16);  // 8 halves (16B)
  return r.v;
}

// B: 32x16 f16 tile from row-major [K][N] matrix, leading dim ldb.
// N = nBase + lane%16; lane<16: halves0-7 -> K=kBase+0..7, halves8-15 -> K=kBase+16..23
// lane>=16: halves0-7 -> K=kBase+8..15, halves8-15 -> K=kBase+24..31
__device__ __forceinline__ v16h frag_b(const _Float16* Bm, int ldb, int kBase,
                                       int nBase, int lane) {
  const _Float16* p = Bm + (size_t)(kBase + ((lane & 16) ? 8 : 0)) * ldb + nBase + (lane & 15);
  v16h v;
#pragma unroll
  for (int i = 0; i < 8; ++i) {
    v[i] = p[(size_t)i * ldb];
    v[8 + i] = p[(size_t)(16 + i) * ldb];
  }
  return v;
}

__device__ __forceinline__ v8f wmma_f16(v16h a, v16h b, v8f c) {
  return __builtin_amdgcn_wmma_f32_16x16x32_f16(false, a, false, b,
                                                (short)0, c, false, false);
}

// ---- staging / setup kernels ----------------------------------------------

// Pack W_amp then W_phase as f16 [2][MP1][C_], zero rows o >= NB_.
__global__ void k_pack_w(const float* __restrict__ Wamp,
                         const float* __restrict__ Wph,
                         _Float16* __restrict__ Wh) {
  int idx = blockIdx.x * blockDim.x + threadIdx.x;
  int n = 2 * MP1 * C_;
  if (idx >= n) return;
  int w = idx / (MP1 * C_);
  int rem = idx - w * (MP1 * C_);
  int o = rem / C_;
  int c = rem - o * C_;
  float v = 0.0f;
  if (o < NB_) v = w ? Wph[o * C_ + c] : Wamp[o * C_ + c];
  Wh[idx] = (_Float16)v;
}

// z f32 -> f16, same [B][C][T] layout.
__global__ void k_pack_z(const float* __restrict__ z,
                         _Float16* __restrict__ zh, int n) {
  int idx = blockIdx.x * blockDim.x + threadIdx.x;
  if (idx < n) zh[idx] = (_Float16)z[idx];
}

// irfft basis F[n][j], f16 [NS_][KP2].
//  j in [0,NB_):       ( s_j/512) * cos(2*pi*j*n/512)
//  j in [NB_,2*NB_):   (-s_k/512) * sin(2*pi*k*n/512), k=j-NB_
//  j >= 514:           0 (K padding)
// scale s_k = 1 for k==0 or k==256, else 2. Exact reduction via (j*n) mod 512.
__global__ void k_basis(_Float16* __restrict__ Fh) {
  int idx = blockIdx.x * blockDim.x + threadIdx.x;
  int tot = NS_ * KP2;
  if (idx >= tot) return;
  int n = idx / KP2;
  int j = idx - n * KP2;
  float v = 0.0f;
  if (j < 2 * NB_) {
    int k = (j < NB_) ? j : (j - NB_);
    int r = (k * n) & (NS_ - 1);                       // exact mod 512
    float ang = (float)r * (6.2831853071795864769f / (float)NS_);
    float sc = ((k == 0) || (k == NS_ / 2)) ? (1.0f / NS_) : (2.0f / NS_);
    v = (j < NB_) ? sc * cosf(ang) : -sc * sinf(ang);
  }
  Fh[idx] = (_Float16)v;
}

// Zero the K-padding rows [2*NB_, KP2) of X[b][KP2][T_].
__global__ void k_zero_pad(_Float16* __restrict__ X) {
  int idx = blockIdx.x * blockDim.x + threadIdx.x;
  int rows = KP2 - 2 * NB_;                            // 30
  int tot = B_ * rows * T_;
  if (idx >= tot) return;
  int b = idx / (rows * T_);
  int rem = idx - b * (rows * T_);
  int r = rem / T_;
  int t = rem - r * T_;
  X[(size_t)b * KP2 * T_ + (size_t)(2 * NB_ + r) * T_ + t] = (_Float16)0.0f;
}

// ---- GEMM1: projections + fused nonlinearity ------------------------------
// One wave -> one 16x16 output tile, two accumulator chains (amp & phase)
// sharing each B fragment. Epilogue: +bias, clip(5).exp, anti-alias mask,
// sincos -> Re/Im rows of X (f16).
__global__ void k_gemm1(const _Float16* __restrict__ Wh,
                        const _Float16* __restrict__ zh,
                        const float* __restrict__ bamp,
                        const float* __restrict__ f0,
                        _Float16* __restrict__ X) {
  const int MT = MP1 / 16;                             // 17
  const int NT = T_ / 16;                              // 125
  int gw = (blockIdx.x * blockDim.x + threadIdx.x) >> 5;
  int lane = threadIdx.x & 31;
  int total = B_ * MT * NT;
  if (gw >= total) return;
  int b = gw / (MT * NT);
  int rem = gw - b * (MT * NT);
  int mt = rem / NT;
  int nt = rem - mt * NT;
  int mBase = mt * 16, nBase = nt * 16;

  const _Float16* WA = Wh;
  const _Float16* WP = Wh + (size_t)MP1 * C_;
  const _Float16* zb = zh + (size_t)b * C_ * T_;

  v8f accA = {};
  v8f accP = {};
#pragma unroll
  for (int k = 0; k < C_ / 32; ++k) {
    int kB = k * 32;
    v16h fb = frag_b(zb, T_, kB, nBase, lane);
    v16h fa = frag_a(WA, C_, mBase, kB, lane);
    accA = wmma_f16(fa, fb, accA);
    v16h fp = frag_a(WP, C_, mBase, kB, lane);
    accP = wmma_f16(fp, fb, accP);
  }

  int n = nBase + (lane & 15);
  int mOff = (lane & 16) ? 8 : 0;
  float f0v = f0[b * T_ + n];
  _Float16* Xb = X + (size_t)b * KP2 * T_;
#pragma unroll
  for (int r = 0; r < 8; ++r) {
    int o = mBase + mOff + r;
    if (o < NB_) {
      float a = accA[r] + bamp[o];
      a = expf(fminf(a, 5.0f));
      if (!((float)o * f0v < (SR_ * 0.5f))) a = 0.0f;  // anti-alias mask
      float s, c;
      sincosf(accP[r], &s, &c);
      Xb[(size_t)o * T_ + n]         = (_Float16)(a * c);  // Re S
      Xb[(size_t)(NB_ + o) * T_ + n] = (_Float16)(a * s);  // Im S
    }
  }
}

// ---- GEMM2: irfft as GEMM -> wt (f32, straight into d_out) ----------------
// One wave -> two 16x16 tiles (M and M+16) sharing each B fragment.
__global__ void k_gemm2(const _Float16* __restrict__ Fh,
                        const _Float16* __restrict__ X,
                        float* __restrict__ wt) {
  const int MPAIR = NS_ / 32;                          // 16
  const int NT = T_ / 16;                              // 125
  int gw = (blockIdx.x * blockDim.x + threadIdx.x) >> 5;
  int lane = threadIdx.x & 31;
  int total = B_ * MPAIR * NT;
  if (gw >= total) return;
  int b = gw / (MPAIR * NT);
  int rem = gw - b * (MPAIR * NT);
  int mp = rem / NT;
  int nt = rem - mp * NT;
  int m0 = mp * 32, nBase = nt * 16;

  const _Float16* Xb = X + (size_t)b * KP2 * T_;
  v8f acc0 = {};
  v8f acc1 = {};
#pragma unroll
  for (int k = 0; k < KP2 / 32; ++k) {
    int kB = k * 32;
    v16h fb = frag_b(Xb, T_, kB, nBase, lane);
    v16h a0 = frag_a(Fh, KP2, m0, kB, lane);
    acc0 = wmma_f16(a0, fb, acc0);
    v16h a1 = frag_a(Fh, KP2, m0 + 16, kB, lane);
    acc1 = wmma_f16(a1, fb, acc1);
  }

  int n = nBase + (lane & 15);
  int mOff = (lane & 16) ? 8 : 0;
  float* wb = wt + (size_t)b * NS_ * T_;
#pragma unroll
  for (int r = 0; r < 8; ++r) {
    int m = m0 + mOff + r;
    wb[(size_t)m * T_ + n]        = acc0[r];
    wb[(size_t)(m + 16) * T_ + n] = acc1[r];
  }
}

// ---- oscillator: per-batch frame-rate phase scan --------------------------
// base[b][t] = sum_{t'<t} HOP * f0[b][t'] / SR  (exclusive prefix)
__global__ void k_scan(const float* __restrict__ f0, float* __restrict__ base) {
  int b = threadIdx.x;
  if (b >= B_) return;
  float acc = 0.0f;
  const float k = (float)HOP / SR_;
  for (int t = 0; t < T_; ++t) {
    base[b * T_ + t] = acc;
    acc += f0[b * T_ + t] * k;
  }
}

// ---- oscillator: per-sample phase, wrap, gather + linear interp -----------
__global__ void k_osc(const float* __restrict__ f0,
                      const float* __restrict__ base,
                      const float* __restrict__ wt,
                      float* __restrict__ harm) {
  int idx = blockIdx.x * blockDim.x + threadIdx.x;
  if (idx >= B_ * L_) return;
  int b = idx / L_;
  int l = idx - b * L_;
  int t = l >> 8;                                      // l / HOP
  int i = l & (HOP - 1);
  float step = f0[b * T_ + t] * (1.0f / SR_);
  float ph = base[b * T_ + t] + (float)(i + 1) * step; // inclusive cumsum
  ph -= floorf(ph);                                    // mod 1
  float pos = ph * (float)NS_;
  float fl = floorf(pos);
  float frac = pos - fl;
  int i0 = ((int)fl) & (NS_ - 1);
  int i1 = (i0 + 1) & (NS_ - 1);
  const float* w = wt + (size_t)b * NS_ * T_ + t;
  float w0 = w[(size_t)i0 * T_];
  float w1 = w[(size_t)i1 * T_];
  harm[idx] = w0 * (1.0f - frac) + w1 * frac;
}

// ---------------------------------------------------------------------------

extern "C" void kernel_launch(void* const* d_in, const int* in_sizes, int n_in,
                              void* d_out, int out_size, void* d_ws,
                              size_t ws_size, hipStream_t stream) {
  const float* z     = (const float*)d_in[0];   // [16,256,2000]
  const float* f0    = (const float*)d_in[1];   // [16,1,2000]
  const float* Wamp  = (const float*)d_in[2];   // [257,256]
  const float* bamp  = (const float*)d_in[3];   // [257]
  const float* Wph   = (const float*)d_in[4];   // [257,256]
  // d_in[5] = sample_rate (24000), d_in[6] = hop_size (256): fixed by setup.

  float* harm = (float*)d_out;                          // [16, 512000]
  float* wt   = (float*)d_out + (size_t)B_ * L_;        // [16, 512, 2000]

  // workspace layout (256B aligned)
  uint8_t* ws = (uint8_t*)d_ws;
  _Float16* Wh   = (_Float16*)(ws + 0);                 // 2*272*256   f16
  _Float16* zh   = (_Float16*)(ws + 278528);            // 16*256*2000 f16
  _Float16* Fh   = (_Float16*)(ws + 16662528);          // 512*544     f16
  _Float16* X    = (_Float16*)(ws + 17219584);          // 16*544*2000 f16
  float*    base = (float*)   (ws + 52035584);          // 16*2000     f32

  const int TPB = 256;

  k_pack_w<<<(2 * MP1 * C_ + TPB - 1) / TPB, TPB, 0, stream>>>(Wamp, Wph, Wh);
  {
    int n = B_ * C_ * T_;
    k_pack_z<<<(n + TPB - 1) / TPB, TPB, 0, stream>>>(z, zh, n);
  }
  k_basis<<<(NS_ * KP2 + TPB - 1) / TPB, TPB, 0, stream>>>(Fh);
  k_zero_pad<<<(B_ * (KP2 - 2 * NB_) * T_ + TPB - 1) / TPB, TPB, 0, stream>>>(X);

  {
    int waves = B_ * (MP1 / 16) * (T_ / 16);            // 34000
    k_gemm1<<<(waves * 32 + TPB - 1) / TPB, TPB, 0, stream>>>(Wh, zh, bamp, f0, X);
  }
  {
    int waves = B_ * (NS_ / 32) * (T_ / 16);            // 32000
    k_gemm2<<<(waves * 32 + TPB - 1) / TPB, TPB, 0, stream>>>(Fh, X, wt);
  }

  k_scan<<<1, 32, 0, stream>>>(f0, base);
  {
    int n = B_ * L_;
    k_osc<<<(n + TPB - 1) / TPB, TPB, 0, stream>>>(f0, base, wt, harm);
  }
}
